// SelfAttention_33689723470288
// MI455X (gfx1250) — matrix operand order repaired and checked by reading, not compiled, not run on hardware
//
#include <hip/hip_runtime.h>
#include <hip/hip_fp16.h>

typedef __attribute__((ext_vector_type(16))) _Float16 v16h;
typedef __attribute__((ext_vector_type(8)))  float    v8f;
typedef __attribute__((ext_vector_type(4)))  float    f32x4;
typedef __attribute__((ext_vector_type(4)))  _Float16 h4;
typedef __attribute__((ext_vector_type(8)))  _Float16 h8;

#define WMMA_F16(a, b, c) \
  __builtin_amdgcn_wmma_f32_16x16x32_f16(false, (a), false, (b), (short)0, (c), false, false)

// ---- CDNA5 async global->LDS copy (ASYNCcnt-tracked, bypasses VGPRs) ----
static __device__ __forceinline__ void async_copy_b128(const void* gptr, void* lds_generic) {
  unsigned lds_off = (unsigned)(size_t)lds_generic;  // generic LDS addr low 32 bits = DS offset
  asm volatile("global_load_async_to_lds_b128 %0, %1, off"
               :: "v"(lds_off), "v"(gptr) : "memory");
}
static __device__ __forceinline__ void wait_async0() {
  asm volatile("s_wait_asynccnt 0" ::: "memory");
}

// ---- cross-lane reductions within the 16-lane halves (matches WMMA C/D layout) ----
static __device__ __forceinline__ float hmax16(float v) {
#pragma unroll
  for (int m = 1; m < 16; m <<= 1) v = fmaxf(v, __shfl_xor(v, m, 32));
  return v;
}
static __device__ __forceinline__ float hsum16(float v) {
#pragma unroll
  for (int m = 1; m < 16; m <<= 1) v += __shfl_xor(v, m, 32);
  return v;
}

// ---------------------------------------------------------------------------
// fp32 -> f16 weight conversion (vectorized)
// ---------------------------------------------------------------------------
__global__ void cvt_f32_f16(const float* __restrict__ s, _Float16* __restrict__ d, int n4) {
  int i = blockIdx.x * 256 + threadIdx.x;
  if (i < n4) {
    f32x4 x = *(const f32x4*)&s[i * 4];
    h4 y;
    y[0] = (_Float16)x[0]; y[1] = (_Float16)x[1];
    y[2] = (_Float16)x[2]; y[3] = (_Float16)x[3];
    *(h4*)&d[i * 4] = y;
  }
}

// ---------------------------------------------------------------------------
// Tiled GEMM  C[M,N] = A[M,K] * W[K,N] + bias, f16 WMMA, fp32 accumulate.
// BM=128 (8 waves x 16 rows), BN=64, BK=32.
// A_F16: A is f16 (async-staged); else fp32 (float4 load -> packed h4 store).
// SPLIT_OUT: write f16 output in [B,H,S,hd] split-head layout, else fp32 row-major.
// ---------------------------------------------------------------------------
template <bool A_F16, bool SPLIT_OUT>
__global__ __launch_bounds__(256) void gemm_wmma(const void* __restrict__ Av,
                                                 const _Float16* __restrict__ W,
                                                 const float* __restrict__ bias,
                                                 void* __restrict__ Cv,
                                                 int M, int N, int K) {
  constexpr int BK = 32;
  __shared__ _Float16 As[128][BK + 8];   // row stride 80B (16B multiple)
  __shared__ _Float16 Ws[BK][64 + 8];    // row stride 144B (16B multiple)

  const int t = threadIdx.x;
  const int w = t >> 5, lane = t & 31, half = lane >> 4, ln = lane & 15;
  const int m0 = blockIdx.y * 128;
  const int n0 = blockIdx.x * 64;

  const float*    Af = (const float*)Av;
  const _Float16* Ah = (const _Float16*)Av;

  v8f acc[4] = {};

  for (int kb = 0; kb < K; kb += BK) {
    // ---- W tile 32x64 f16: async copy, one b128 chunk / thread ----
    {
      int r = t >> 3, c8 = (t & 7) * 8;
      async_copy_b128(&W[(size_t)(kb + r) * N + n0 + c8], &Ws[r][c8]);
    }
    // ---- A tile 128x32 ----
    if constexpr (A_F16) {
#pragma unroll
      for (int i = 0; i < 2; ++i) {      // 512 b128 chunks, 2 / thread
        int e = t + i * 256, r = e >> 2, c8 = (e & 3) * 8;
        async_copy_b128(&Ah[(size_t)(m0 + r) * K + kb + c8], &As[r][c8]);
      }
    } else {
#pragma unroll
      for (int i = 0; i < 4; ++i) {      // float4 load -> packed h4 store
        int e = t + i * 256, r = e >> 3, c4 = (e & 7) * 4;
        f32x4 x = *(const f32x4*)&Af[(size_t)(m0 + r) * K + kb + c4];
        h4 y;
        y[0] = (_Float16)x[0]; y[1] = (_Float16)x[1];
        y[2] = (_Float16)x[2]; y[3] = (_Float16)x[3];
        *(h4*)&As[r][c4] = y;
      }
    }
    wait_async0();
    __syncthreads();

    // A fragment 16x32: M = lane&15; K = 0..15 (+16 for upper half-wave)
    v16h af;
#pragma unroll
    for (int i = 0; i < 16; ++i)
      af[i] = As[w * 16 + ln][i + half * 16];

#pragma unroll
    for (int g = 0; g < 4; ++g) {   // B fragment 32x16: N = lane&15, K over VGPRs
      v16h bf;
#pragma unroll
      for (int i = 0; i < 16; ++i)
        bf[i] = Ws[half * 16 + i][g * 16 + ln];
      acc[g] = WMMA_F16(af, bf, acc[g]);
    }
    __syncthreads();
  }

#pragma unroll
  for (int g = 0; g < 4; ++g)
#pragma unroll
    for (int r = 0; r < 8; ++r) {
      int row = m0 + w * 16 + r + half * 8;     // C/D layout: VGPR r, halves = M / M+8
      int col = n0 + g * 16 + ln;
      float c = acc[g][r] + bias[col];
      if (SPLIT_OUT) {  // [B,S,D] -> [B,H,S,hd] f16
        int b = row >> 10, s = row & 1023, h = col >> 6, d = col & 63;
        ((_Float16*)Cv)[(((size_t)(b * 16 + h)) * 1024 + s) * 64 + d] = (_Float16)c;
      } else {
        ((float*)Cv)[(size_t)row * N + col] = c;
      }
    }
}

// ---------------------------------------------------------------------------
// Masked attention: per WG one (b,h) and 128 query rows (8 waves x 16 rows).
// Pass 1: online row-max m and temperature denom Zt = sum exp((s-m)/TEMP).
// Pass 2: recompute scores; keep iff exp((s-m)*1000) >= THRESH*Zt;
//         accumulate p=exp(s-m), O += P*V (WMMA), normalize by sum(p).
// K/V tiles are staged with CDNA5 async global->LDS copies.
// ---------------------------------------------------------------------------
__global__ __launch_bounds__(256) void attn_masked(const _Float16* __restrict__ qh,
                                                   const _Float16* __restrict__ kh,
                                                   const _Float16* __restrict__ vh,
                                                   _Float16* __restrict__ aout) {
  constexpr int S = 1024;
  __shared__ _Float16 kts[64][64 + 8];    // row stride 144B
  __shared__ _Float16 vts[64][64 + 8];
  __shared__ _Float16 ps[8][16][64 + 8];  // per-wave P scratch (C-layout -> A-layout)

  const int t = threadIdx.x;
  const int w = t >> 5, lane = t & 31, half = lane >> 4, ln = lane & 15;
  const int bh = blockIdx.y, b = bh >> 4, h = bh & 15;
  const size_t base = (size_t)bh * S * 64;
  const int q0 = blockIdx.x * 128 + w * 16;

  // q fragments (hd=64 -> two K=32 chunks), resident for the whole kernel
  v16h qa[2];
#pragma unroll
  for (int kc = 0; kc < 2; ++kc)
#pragma unroll
    for (int i = 0; i < 16; ++i) {
      int d = kc * 32 + half * 16 + i;
      qa[kc][i] = qh[base + (size_t)(q0 + ln) * 64 + d];
    }

  float m8[8], zt[8];
#pragma unroll
  for (int r = 0; r < 8; ++r) { m8[r] = -3.0e38f; zt[r] = 0.f; }

  // ---------------- pass 1 ----------------
  for (int kt = 0; kt < S / 64; ++kt) {
    __builtin_prefetch(kh + base + (size_t)((kt + 1) & 15) * 64 * 64, 0, 0);
#pragma unroll
    for (int i = 0; i < 2; ++i) {   // 64x64 f16 tile = 512 b128 chunks, 2 / thread
      int e = t + i * 256, r = e >> 3, c8 = (e & 7) * 8;
      async_copy_b128(&kh[base + (size_t)(kt * 64 + r) * 64 + c8], &kts[r][c8]);
    }
    wait_async0();
    __syncthreads();

    v8f sc[4] = {};
#pragma unroll
    for (int g = 0; g < 4; ++g)
#pragma unroll
      for (int kc = 0; kc < 2; ++kc) {
        v16h bf;  // B = K^T: N = k-column of tile, contraction = d (contiguous in LDS)
#pragma unroll
        for (int i = 0; i < 16; ++i)
          bf[i] = kts[g * 16 + ln][kc * 32 + half * 16 + i];
        sc[g] = WMMA_F16(qa[kc], bf, sc[g]);
      }
#pragma unroll
    for (int g = 0; g < 4; ++g) sc[g] *= 0.125f;  // 1/sqrt(64)

#pragma unroll
    for (int r = 0; r < 8; ++r) {
      float tm = fmaxf(fmaxf(sc[0][r], sc[1][r]), fmaxf(sc[2][r], sc[3][r]));
      tm = hmax16(tm);
      float nm = fmaxf(m8[r], tm);
      float part = 0.f;
#pragma unroll
      for (int g = 0; g < 4; ++g) part += __expf((sc[g][r] - nm) * 1000.f);
      part = hsum16(part);
      zt[r] = zt[r] * __expf((m8[r] - nm) * 1000.f) + part;
      m8[r] = nm;
    }
    __syncthreads();
  }

  // ---------------- pass 2 ----------------
  float l2[8] = {};
  v8f oacc[4] = {};
  for (int kt = 0; kt < S / 64; ++kt) {
#pragma unroll
    for (int i = 0; i < 2; ++i) {
      int e = t + i * 256, r = e >> 3, c8 = (e & 7) * 8;
      async_copy_b128(&kh[base + (size_t)(kt * 64 + r) * 64 + c8], &kts[r][c8]);
      async_copy_b128(&vh[base + (size_t)(kt * 64 + r) * 64 + c8], &vts[r][c8]);
    }
    wait_async0();
    __syncthreads();

    v8f sc[4] = {};
#pragma unroll
    for (int g = 0; g < 4; ++g)
#pragma unroll
      for (int kc = 0; kc < 2; ++kc) {
        v16h bf;
#pragma unroll
        for (int i = 0; i < 16; ++i)
          bf[i] = kts[g * 16 + ln][kc * 32 + half * 16 + i];
        sc[g] = WMMA_F16(qa[kc], bf, sc[g]);
      }
#pragma unroll
    for (int g = 0; g < 4; ++g) sc[g] *= 0.125f;

    // data-dependent mask + unnormalized softmax numerators
#pragma unroll
    for (int g = 0; g < 4; ++g)
#pragma unroll
      for (int r = 0; r < 8; ++r) {
        float s = sc[g][r];
        float am = __expf((s - m8[r]) * 1000.f);            // att_map * Zt
        float p = (am >= 0.019f * zt[r]) ? __expf(s - m8[r]) : 0.f;
        l2[r] += p;
        ps[w][r + half * 8][g * 16 + ln] = (_Float16)p;
      }
    asm volatile("s_wait_dscnt 0" ::: "memory");  // wave-private LDS RAW

    v16h pa[2];  // P as A fragments (contraction over 64 k-columns)
#pragma unroll
    for (int kc = 0; kc < 2; ++kc)
#pragma unroll
      for (int i = 0; i < 16; ++i)
        pa[kc][i] = ps[w][ln][kc * 32 + half * 16 + i];

#pragma unroll
    for (int og = 0; og < 4; ++og)
#pragma unroll
      for (int kc = 0; kc < 2; ++kc) {
        v16h bf;  // B = V tile, row-major: N = output dim
#pragma unroll
        for (int i = 0; i < 16; ++i)
          bf[i] = vts[kc * 32 + half * 16 + i][og * 16 + ln];
        oacc[og] = WMMA_F16(pa[kc], bf, oacc[og]);
      }
    __syncthreads();
  }

#pragma unroll
  for (int r = 0; r < 8; ++r) l2[r] = hsum16(l2[r]);

#pragma unroll
  for (int og = 0; og < 4; ++og)
#pragma unroll
    for (int r = 0; r < 8; ++r) {
      int srow = q0 + r + half * 8;
      int col  = h * 64 + og * 16 + ln;  // merge heads back to [B,S,D]
      float o = oacc[og][r] / fmaxf(l2[r], 1e-20f);
      aout[((size_t)b * 1024 + srow) * 1024 + col] = (_Float16)o;
    }
}

// ---------------------------------------------------------------------------
extern "C" void kernel_launch(void* const* d_in, const int* in_sizes, int n_in,
                              void* d_out, int out_size, void* d_ws, size_t ws_size,
                              hipStream_t stream) {
  const float* Q  = (const float*)d_in[0];
  const float* Kx = (const float*)d_in[1];
  const float* V  = (const float*)d_in[2];
  const float* Wq = (const float*)d_in[3];
  const float* bq = (const float*)d_in[4];
  const float* Wk = (const float*)d_in[5];
  const float* bk = (const float*)d_in[6];
  const float* Wv = (const float*)d_in[7];
  const float* bv = (const float*)d_in[8];
  const float* Wo = (const float*)d_in[9];
  const float* bo = (const float*)d_in[10];

  // workspace layout (needs 40 MB): 4x f16 weights, q/k/v heads f16, attn_out f16
  char* ws = (char*)d_ws;
  const size_t MB = 1024u * 1024u;
  _Float16* Wqh = (_Float16*)(ws + 0 * MB);
  _Float16* Wkh = (_Float16*)(ws + 2 * MB);
  _Float16* Wvh = (_Float16*)(ws + 4 * MB);
  _Float16* Woh = (_Float16*)(ws + 6 * MB);
  _Float16* qhp = (_Float16*)(ws + 8 * MB);
  _Float16* khp = (_Float16*)(ws + 16 * MB);
  _Float16* vhp = (_Float16*)(ws + 24 * MB);
  _Float16* aoh = (_Float16*)(ws + 32 * MB);

  const int D = 1024, Mrows = 4096;
  const int n4 = D * D / 4;
  cvt_f32_f16<<<(n4 + 255) / 256, 256, 0, stream>>>(Wq, Wqh, n4);
  cvt_f32_f16<<<(n4 + 255) / 256, 256, 0, stream>>>(Wk, Wkh, n4);
  cvt_f32_f16<<<(n4 + 255) / 256, 256, 0, stream>>>(Wv, Wvh, n4);
  cvt_f32_f16<<<(n4 + 255) / 256, 256, 0, stream>>>(Wo, Woh, n4);

  dim3 gg(D / 64, Mrows / 128);  // 16 x 32 workgroups
  gemm_wmma<false, true><<<gg, 256, 0, stream>>>(Q,  Wqh, bq, qhp, Mrows, D, D);
  gemm_wmma<false, true><<<gg, 256, 0, stream>>>(Kx, Wkh, bk, khp, Mrows, D, D);
  gemm_wmma<false, true><<<gg, 256, 0, stream>>>(V,  Wvh, bv, vhp, Mrows, D, D);

  attn_masked<<<dim3(8, 64), 256, 0, stream>>>(qhp, khp, vhp, aoh);

  gemm_wmma<true, false><<<gg, 256, 0, stream>>>(aoh, Woh, bo, d_out, Mrows, D, D);
}
